// MessageLayer_35948876267548
// MI455X (gfx1250) — matrix operand order repaired and checked by reading, not compiled, not run on hardware
//
#include <hip/hip_runtime.h>

typedef __attribute__((ext_vector_type(16))) __bf16 v16bf;
typedef __attribute__((ext_vector_type(8)))  float  v8f;
typedef __attribute__((ext_vector_type(8)))  short  short8;

union Frag { v16bf v; short8 h[2]; };

__device__ inline unsigned short f2bf(float f) {
    unsigned u = __float_as_uint(f);
    unsigned r = (u + 0x7FFFu + ((u >> 16) & 1u)) >> 16;   // round-to-nearest-even
    return (unsigned short)r;
}
__device__ inline float bf2f(unsigned short b) {
    return __uint_as_float(((unsigned)b) << 16);
}

__device__ inline void atomicMaxF(float* addr, float val) {
    unsigned* ua = (unsigned*)addr;
    unsigned old = *ua;
    while (true) {
        float f = __uint_as_float(old);
        if (f >= val) break;
        unsigned prev = atomicCAS(ua, old, __float_as_uint(val));
        if (prev == old) break;
        old = prev;
    }
}

// A fragment (16x32 bf16): lane L<16 holds row L, K in {k0..k0+7} U {k0+16..k0+23};
// lanes 16..31 hold the +8 K offsets. Two contiguous 16B LDS chunks per lane.
__device__ inline v16bf load_a_lds(const unsigned short* act, int W, int k0, int lane) {
    int r = lane & 15, hi = lane >> 4;
    const unsigned short* p = act + r * W + k0 + 8 * hi;
    Frag f;
    f.h[0] = *(const short8*)p;
    f.h[1] = *(const short8*)(p + 16);
    return f.v;
}

// B fragment (32x16 bf16, K x N): lane L holds column N = L%16, K in {8*(L/16)..+7} U {16+8*(L/16)..+7}.
// Weights stored transposed [out][in] so each lane reads two 16-byte runs along K, 32 bytes apart.
__device__ inline v16bf load_b_glb(const unsigned short* wt, int in_w, int n0, int k0, int lane) {
    int n = lane & 15, hi = lane >> 4;
    const unsigned short* p = wt + (size_t)(n0 + n) * in_w + k0 + 8 * hi;
    Frag f;
    f.h[0] = *(const short8*)p;
    f.h[1] = *(const short8*)(p + 16);
    return f.v;
}

// One dense layer over a 32-row tile: two 16-row sub-tiles share every B fragment,
// so each weight fetch feeds two WMMAs (two independent accumulation chains).
template <int IN_W, int OUT_W, bool RELU>
__device__ inline void mlp_layer2(const unsigned short* in_lds, unsigned short* out_lds,
                                  const unsigned short* wt, const float* bias, int lane) {
    int n = lane & 15, rbase = 8 * (lane >> 4);
#pragma unroll 1
    for (int n0 = 0; n0 < OUT_W; n0 += 16) {
        v8f acc0 = {}, acc1 = {};
#pragma unroll
        for (int k0 = 0; k0 < IN_W; k0 += 32) {
            v16bf b  = load_b_glb(wt, IN_W, n0, k0, lane);
            v16bf a0 = load_a_lds(in_lds, IN_W, k0, lane);
            v16bf a1 = load_a_lds(in_lds + 16 * IN_W, IN_W, k0, lane);
            acc0 = __builtin_amdgcn_wmma_f32_16x16x32_bf16(false, a0, false, b,
                                                           (short)0, acc0, false, false);
            acc1 = __builtin_amdgcn_wmma_f32_16x16x32_bf16(false, a1, false, b,
                                                           (short)0, acc1, false, false);
        }
        float bv = bias[n0 + n];
#pragma unroll
        for (int i = 0; i < 8; ++i) {
            float v0 = acc0[i] + bv;
            float v1 = acc1[i] + bv;
            if (RELU) { v0 = fmaxf(v0, 0.0f); v1 = fmaxf(v1, 0.0f); }
            out_lds[(rbase + i) * OUT_W + n0 + n]        = f2bf(v0);
            out_lds[(rbase + i + 16) * OUT_W + n0 + n]   = f2bf(v1);
        }
    }
}

__global__ void init_kernel(const float* __restrict__ atom_fea, float* __restrict__ out,
                            float* __restrict__ gmax, float* __restrict__ denom, int N) {
    long i = (long)blockIdx.x * blockDim.x + threadIdx.x;
    if (i < (long)N * 64) out[i] = atom_fea[i];
    if (i < 3L * N) { gmax[i] = -3.0e38f; denom[i] = 0.0f; }
}

__global__ void gather_kernel(const float* __restrict__ atom_fea, const float* __restrict__ atom_w,
                              const int* __restrict__ self_idx, const int* __restrict__ nbr_idx,
                              unsigned short* __restrict__ fea, float* __restrict__ wnbr, int M) {
    int e = blockIdx.x * blockDim.x + threadIdx.x;
    if (e >= M) return;
    int s = self_idx[e], nb = nbr_idx[e];
    const float* ps = atom_fea + (size_t)s * 64;
    const float* pn = atom_fea + (size_t)nb * 64;
    unsigned short* dst = fea + (size_t)e * 128;
#pragma unroll 4
    for (int d = 0; d < 64; ++d) dst[d] = f2bf(ps[d]);
#pragma unroll 4
    for (int d = 0; d < 64; ++d) dst[64 + d] = f2bf(pn[d]);
    wnbr[e] = atom_w[nb];
}

// src [H][fi][fo] fp32 -> dst [H][fo][fi] bf16
__global__ void transpose_w_kernel(const float* __restrict__ src, unsigned short* __restrict__ dst,
                                   int fi, int fo, long total) {
    long i = (long)blockIdx.x * blockDim.x + threadIdx.x;
    if (i >= total) return;
    int o  = (int)(i % fo);
    long t = i / fo;
    int ii = (int)(t % fi);
    int h  = (int)(t / fi);
    dst[((size_t)h * fo + o) * fi + ii] = f2bf(src[i]);
}

// ---- fused gate MLP: 128 -> 384 -> 256 -> 64 -> 1, one wave per 32-edge tile ----
__global__ __launch_bounds__(32) void gate_kernel(
    const unsigned short* __restrict__ fea,
    const unsigned short* __restrict__ w1, const unsigned short* __restrict__ w2,
    const unsigned short* __restrict__ w3, const unsigned short* __restrict__ w4,
    const float* __restrict__ b1, const float* __restrict__ b2,
    const float* __restrict__ b3, const float* __restrict__ b4,
    const int* __restrict__ self_idx,
    float* __restrict__ gate_out, float* __restrict__ gmax, int M) {
    __shared__ __align__(16) unsigned short bufA[32 * 384];
    __shared__ __align__(16) unsigned short bufB[32 * 384];
    int lane = threadIdx.x & 31;
    int h = blockIdx.y;
    int tile = blockIdx.x;                 // 32-edge tile
    // stage 32x128 bf16 edge-feature tile into LDS (zero-pad past M)
    for (int idx = lane; idx < 32 * 16; idx += 32) {
        int r = idx >> 4, c = (idx & 15) * 8;
        int e = tile * 32 + r;
        short8 val = {};
        if (e < M) val = *(const short8*)(fea + (size_t)e * 128 + c);
        *(short8*)(bufA + r * 128 + c) = val;
    }
    mlp_layer2<128, 384, true>(bufA, bufB, w1 + (size_t)h * 128 * 384, b1 + h * 384, lane);
    mlp_layer2<384, 256, true>(bufB, bufA, w2 + (size_t)h * 384 * 256, b2 + h * 256, lane);
    mlp_layer2<256, 64,  true>(bufA, bufB, w3 + (size_t)h * 256 * 64,  b3 + h * 64,  lane);
    // final 64 -> 1: one edge per lane (all 32 lanes busy)
    {
        int e = tile * 32 + lane;
        if (e < M) {
            const unsigned short* wv = w4 + h * 64;
            float g = b4[h];
#pragma unroll
            for (int k = 0; k < 64; ++k)
                g += bf2f(bufB[lane * 64 + k]) * bf2f(wv[k]);
            gate_out[(size_t)h * M + e] = g;
            atomicMaxF(&gmax[(size_t)self_idx[e] * 3 + h], g);
        }
    }
}

__global__ void exp_kernel(const float* __restrict__ gate, const float* __restrict__ gmax,
                           const float* __restrict__ wnbr, const int* __restrict__ self_idx,
                           float* __restrict__ evals, float* __restrict__ denom, int M) {
    long i = (long)blockIdx.x * blockDim.x + threadIdx.x;
    if (i >= 3L * M) return;
    int e = (int)(i % M);
    int h = (int)(i / M);
    int s = self_idx[e];
    float ev = wnbr[e] * __expf(gate[i] - gmax[(size_t)s * 3 + h]);
    evals[i] = ev;
    atomicAdd(&denom[(size_t)s * 3 + h], ev);
}

// ---- fused message MLP: 128 -> 384 -> 256 -> 128 -> 64, scaled scatter-add ----
__global__ __launch_bounds__(32) void msg_kernel(
    const unsigned short* __restrict__ fea,
    const unsigned short* __restrict__ w1, const unsigned short* __restrict__ w2,
    const unsigned short* __restrict__ w3, const unsigned short* __restrict__ w4,
    const float* __restrict__ b1, const float* __restrict__ b2,
    const float* __restrict__ b3, const float* __restrict__ b4,
    const int* __restrict__ self_idx,
    const float* __restrict__ evals, const float* __restrict__ denom,
    float* __restrict__ out, int M) {
    __shared__ __align__(16) unsigned short bufA[32 * 384];
    __shared__ __align__(16) unsigned short bufB[32 * 384];
    __shared__ float coefs[32];
    int lane = threadIdx.x & 31;
    int h = blockIdx.y;
    int tile = blockIdx.x;
    for (int idx = lane; idx < 32 * 16; idx += 32) {
        int r = idx >> 4, c = (idx & 15) * 8;
        int e = tile * 32 + r;
        short8 val = {};
        if (e < M) val = *(const short8*)(fea + (size_t)e * 128 + c);
        *(short8*)(bufA + r * 128 + c) = val;
    }
    mlp_layer2<128, 384, true>(bufA, bufB, w1 + (size_t)h * 128 * 384, b1 + h * 384, lane);
    mlp_layer2<384, 256, true>(bufB, bufA, w2 + (size_t)h * 384 * 256, b2 + h * 256, lane);
    mlp_layer2<256, 128, true>(bufA, bufB, w3 + (size_t)h * 256 * 128, b3 + h * 128, lane);
    // per-row softmax coefficient (includes 1/H for the head mean), one row per lane
    {
        int e = tile * 32 + lane;
        float c = 0.0f;
        if (e < M) {
            int s = self_idx[e];
            c = evals[(size_t)h * M + e] / (denom[(size_t)s * 3 + h] + 1e-13f) * (1.0f / 3.0f);
        }
        coefs[lane] = c;
    }
    // final linear 128 -> 64 in WMMA (B shared by both row sub-tiles), scale + scatter
    int n = lane & 15, rbase = 8 * (lane >> 4);
    const unsigned short* w4h = w4 + (size_t)h * 128 * 64;
#pragma unroll 1
    for (int n0 = 0; n0 < 64; n0 += 16) {
        v8f acc0 = {}, acc1 = {};
#pragma unroll
        for (int k0 = 0; k0 < 128; k0 += 32) {
            v16bf b  = load_b_glb(w4h, 128, n0, k0, lane);
            v16bf a0 = load_a_lds(bufB, 128, k0, lane);
            v16bf a1 = load_a_lds(bufB + 16 * 128, 128, k0, lane);
            acc0 = __builtin_amdgcn_wmma_f32_16x16x32_bf16(false, a0, false, b,
                                                           (short)0, acc0, false, false);
            acc1 = __builtin_amdgcn_wmma_f32_16x16x32_bf16(false, a1, false, b,
                                                           (short)0, acc1, false, false);
        }
        float bv = b4[h * 64 + n0 + n];
#pragma unroll
        for (int i = 0; i < 8; ++i) {
            int m0 = rbase + i;
            int e0 = tile * 32 + m0;
            if (e0 < M) {
                float v = (acc0[i] + bv) * coefs[m0];
                atomicAdd(&out[(size_t)self_idx[e0] * 64 + n0 + n], v);
            }
            int m1 = m0 + 16;
            int e1 = tile * 32 + m1;
            if (e1 < M) {
                float v = (acc1[i] + bv) * coefs[m1];
                atomicAdd(&out[(size_t)self_idx[e1] * 64 + n0 + n], v);
            }
        }
    }
}

extern "C" void kernel_launch(void* const* d_in, const int* in_sizes, int n_in,
                              void* d_out, int out_size, void* d_ws, size_t ws_size,
                              hipStream_t stream) {
    const float* atom_w   = (const float*)d_in[0];
    const float* atom_fea = (const float*)d_in[1];
    const int*   self_idx = (const int*)d_in[2];
    const int*   nbr_idx  = (const int*)d_in[3];
    const float* gW1 = (const float*)d_in[4];  const float* gb1 = (const float*)d_in[5];
    const float* gW2 = (const float*)d_in[6];  const float* gb2 = (const float*)d_in[7];
    const float* gW3 = (const float*)d_in[8];  const float* gb3 = (const float*)d_in[9];
    const float* gW4 = (const float*)d_in[10]; const float* gb4 = (const float*)d_in[11];
    const float* mW1 = (const float*)d_in[12]; const float* mb1 = (const float*)d_in[13];
    const float* mW2 = (const float*)d_in[14]; const float* mb2 = (const float*)d_in[15];
    const float* mW3 = (const float*)d_in[16]; const float* mb3 = (const float*)d_in[17];
    const float* mW4 = (const float*)d_in[18]; const float* mb4 = (const float*)d_in[19];
    (void)n_in; (void)out_size; (void)ws_size;

    const int M = in_sizes[2];
    const int N = in_sizes[1] / 64;
    float* out = (float*)d_out;

    char* ws = (char*)d_ws;
    size_t off = 0;
    auto carve = [&](size_t bytes) -> void* {
        void* p = ws + off;
        off = (off + bytes + 255) & ~(size_t)255;
        return p;
    };
    unsigned short* fea  = (unsigned short*)carve((size_t)M * 128 * 2);
    float* wnbr  = (float*)carve((size_t)M * 4);
    float* gate  = (float*)carve(3ull * M * 4);
    float* evals = (float*)carve(3ull * M * 4);
    float* gmax  = (float*)carve(3ull * N * 4);
    float* denom = (float*)carve(3ull * N * 4);
    unsigned short* wtG1 = (unsigned short*)carve(3ull * 128 * 384 * 2);
    unsigned short* wtG2 = (unsigned short*)carve(3ull * 384 * 256 * 2);
    unsigned short* wtG3 = (unsigned short*)carve(3ull * 256 * 64 * 2);
    unsigned short* wtG4 = (unsigned short*)carve(3ull * 64 * 2);
    unsigned short* wtM1 = (unsigned short*)carve(3ull * 128 * 384 * 2);
    unsigned short* wtM2 = (unsigned short*)carve(3ull * 384 * 256 * 2);
    unsigned short* wtM3 = (unsigned short*)carve(3ull * 256 * 128 * 2);
    unsigned short* wtM4 = (unsigned short*)carve(3ull * 128 * 64 * 2);

    long initTot = (long)N * 64;
    init_kernel<<<(unsigned)((initTot + 255) / 256), 256, 0, stream>>>(atom_fea, out, gmax, denom, N);
    gather_kernel<<<(unsigned)((M + 127) / 128), 128, 0, stream>>>(atom_fea, atom_w, self_idx,
                                                                   nbr_idx, fea, wnbr, M);
    auto T = [&](const float* s, unsigned short* d, int fi, int fo) {
        long tot = 3L * fi * fo;
        transpose_w_kernel<<<(unsigned)((tot + 255) / 256), 256, 0, stream>>>(s, d, fi, fo, tot);
    };
    T(gW1, wtG1, 128, 384); T(gW2, wtG2, 384, 256); T(gW3, wtG3, 256, 64); T(gW4, wtG4, 64, 1);
    T(mW1, wtM1, 128, 384); T(mW2, wtM2, 384, 256); T(mW3, wtM3, 256, 128); T(mW4, wtM4, 128, 64);

    int nTiles32 = (M + 31) / 32;
    dim3 grid((unsigned)nTiles32, 3);
    gate_kernel<<<grid, 32, 0, stream>>>(fea, wtG1, wtG2, wtG3, wtG4,
                                         gb1, gb2, gb3, gb4, self_idx, gate, gmax, M);
    long eTot = 3L * M;
    exp_kernel<<<(unsigned)((eTot + 255) / 256), 256, 0, stream>>>(gate, gmax, wnbr, self_idx,
                                                                   evals, denom, M);
    msg_kernel<<<grid, 32, 0, stream>>>(fea, wtM1, wtM2, wtM3, wtM4,
                                        mb1, mb2, mb3, mb4, self_idx, evals, denom, out, M);
}
